// GAT_Block_64922725647075
// MI455X (gfx1250) — matrix operand order repaired and checked by reading, not compiled, not run on hardware
//
#include <hip/hip_runtime.h>

// ---------------------------------------------------------------------------
// Fused GAT attention block for MI455X (gfx1250, wave32).
//   x:(16,3,512,64)  mask:(64,64)  W1:(3,64) b1:(64) W2:(32,1) b2:(1)
//   out:(16,64,512,64)
// One 64-thread block (2 waves) handles one (batch, frame, head-pair).
// Each wave owns one head: h -> s_src/s_dst -> lrelu/mask/softmax ->
// attn(64x64) @ h(64x16) via v_wmma_f32_16x16x32_f16.
// ---------------------------------------------------------------------------

typedef __attribute__((ext_vector_type(16))) _Float16 v16h;
typedef __attribute__((ext_vector_type(8)))  float    v8f;

#define NNODE 64
#define HD    16
#define NH    4
#define HPB   2          // heads per block (one per wave)
#define F_DIM 512
#define C_IN  3
#define LRELU 0.01f

union F16Frag { v16h v; unsigned int u[8]; };

__launch_bounds__(64)
__global__ void gat_fused_kernel(const float* __restrict__ x,
                                 const float* __restrict__ mask,
                                 const float* __restrict__ W1,
                                 const float* __restrict__ b1,
                                 const float* __restrict__ W2,
                                 const float* __restrict__ b2,
                                 float* __restrict__ out)
{
    // ---- LDS (~49 KB total, all row strides padded to kill bank conflicts)
    __shared__ float sx[C_IN * NNODE];              // x rows for this (b,f)
    __shared__ float sw1[C_IN * HPB * HD];          // 3 x 32 slice of W1
    __shared__ float sb1[HPB * HD];
    __shared__ float sw2[2 * HD];                   // a_src | a_dst
    __shared__ float smask[NNODE * 65];             // padded 64x64 mask
    __shared__ alignas(16) _Float16 shT[HPB][HD * 66];      // h^T  (d,k) f16
    __shared__ alignas(16) _Float16 sattn[HPB][NNODE * 66]; // exp(scores) f16
    __shared__ float ssrc[HPB * NNODE];
    __shared__ float sdst[HPB * NNODE];
    __shared__ float srinv[HPB * NNODE];            // 1/rowsum
    __shared__ float sout[HPB][HD * 65];            // out^T staging (d,n)

    const int t  = threadIdx.x;      // 0..63
    const int w  = t >> 5;           // wave id == local head
    const int l  = t & 31;           // lane
    const int f  = blockIdx.x;       // frame 0..511
    const int b  = blockIdx.y;       // batch 0..15
    const int hp = blockIdx.z;       // head pair 0..1
    const float b2v = b2[0];

    // ---- stage 0: cooperative loads ------------------------------------
    #pragma unroll
    for (int i = t; i < C_IN * NNODE; i += 64) {
        int c = i >> 6, n = i & 63;
        sx[i] = x[(((size_t)b * C_IN + c) * F_DIM + f) * NNODE + n];
    }
    #pragma unroll
    for (int i = t; i < C_IN * HPB * HD; i += 64) {
        int c = i >> 5, jj = i & 31;
        sw1[i] = W1[c * (NH * HD) + hp * (HPB * HD) + jj];
    }
    if (t < HPB * HD) sb1[t] = b1[hp * (HPB * HD) + t];
    if (t < 2 * HD)   sw2[t] = W2[t];
    for (int i = t; i < NNODE * NNODE; i += 64) {
        int r = i >> 6, c = i & 63;
        smask[r * 65 + c] = mask[i];
    }
    __syncthreads();

    // ---- stage 1: h = xp @ W1 + b1  (store transposed f16 for WMMA B) --
    for (int i = t; i < NNODE * HPB * HD; i += 64) {
        int jj = i & 31;             // local feature (head*16+d)
        int n  = i >> 5;             // node
        float v = sx[n]        * sw1[jj]
                + sx[64 + n]   * sw1[32 + jj]
                + sx[128 + n]  * sw1[64 + jj]
                + sb1[jj];
        shT[jj >> 4][(jj & 15) * 66 + n] = (_Float16)v;
    }
    __syncthreads();

    // ---- stage 2: per-head attention logits ----------------------------
    #pragma unroll
    for (int rr = 0; rr < 2; ++rr) {
        int n = l + rr * 32;
        float a = 0.f, d = 0.f;
        #pragma unroll
        for (int k = 0; k < HD; ++k) {
            float hv = (float)shT[w][k * 66 + n];
            a += hv * sw2[k];
            d += hv * sw2[HD + k];
        }
        ssrc[w * NNODE + n] = a;
        sdst[w * NNODE + n] = d;
    }
    __syncthreads();

    // ---- stage 3: lrelu, mask, softmax (unnormalized exp -> f16) -------
    #pragma unroll
    for (int rr = 0; rr < 2; ++rr) {
        int i = l + rr * 32;
        float si = ssrc[w * NNODE + i];
        float m = -3.0e38f;
        for (int j = 0; j < NNODE; ++j) {
            float sc = si + sdst[w * NNODE + j] + b2v;
            sc = (sc > 0.f) ? sc : LRELU * sc;
            sc *= smask[i * 65 + j];
            m = fmaxf(m, sc);
        }
        float sum = 0.f;
        for (int j = 0; j < NNODE; ++j) {
            float sc = si + sdst[w * NNODE + j] + b2v;
            sc = (sc > 0.f) ? sc : LRELU * sc;
            sc *= smask[i * 65 + j];
            float e = __expf(sc - m);
            sum += e;
            sattn[w][i * 66 + j] = (_Float16)e;
        }
        srinv[w * NNODE + i] = 1.0f / sum;
    }
    __syncthreads();

    // ---- stage 4: out = softmax @ h  via v_wmma_f32_16x16x32_f16 -------
    {
        const int Nc = l & 15;            // column (= d for B/D)
        const int hi = (l >> 4) & 1;      // lane half selector
        #pragma unroll
        for (int tile = 0; tile < 4; ++tile) {      // 4 row tiles of 16
            v8f acc = {};
            #pragma unroll
            for (int kc = 0; kc < 2; ++kc) {        // K = 64 in 2x32 chunks
                F16Frag A, Bf;
                // A: 16x32 f16, lane&15 = M row; lanes 0-15 K=0..15, 16-31 K=16..31
                int M   = tile * 16 + Nc;
                int kbA = kc * 32 + hi * 16;
                const _Float16* ar = &sattn[w][M * 66];
                #pragma unroll
                for (int v = 0; v < 8; ++v)
                    A.u[v] = *(const unsigned int*)&ar[kbA + 2 * v];
                // B: 32x16 f16, lane&15 = N col; lanes 0-15 K=0..15, 16-31 K=16..31
                int kbB = kc * 32 + hi * 16;
                const _Float16* br = &shT[w][Nc * 66];
                #pragma unroll
                for (int v = 0; v < 8; ++v)
                    Bf.u[v] = *(const unsigned int*)&br[kbB + 2 * v];
                acc = __builtin_amdgcn_wmma_f32_16x16x32_f16(
                        false, A.v, false, Bf.v, (short)0, acc, false, false);
            }
            // D layout: VGPR r -> M = r (+8 for upper lanes); scale by 1/rowsum
            int mb = tile * 16 + hi * 8;
            #pragma unroll
            for (int r = 0; r < 8; ++r) {
                int M = mb + r;
                sout[w][Nc * 65 + M] = acc[r] * srinv[w * NNODE + M];
            }
        }
    }
    __syncthreads();

    // ---- stage 5: coalesced store into out(b, 64, f, n) ----------------
    for (int i = t; i < HPB * HD * NNODE; i += 64) {
        int jl = i >> 6;                 // local feature 0..31
        int n  = i & 63;
        int j  = hp * (HPB * HD) + jl;   // global feature
        out[(((size_t)b * (NH * HD) + j) * F_DIM + f) * NNODE + n] =
            sout[jl >> 4][(jl & 15) * 65 + n];
    }
}

extern "C" void kernel_launch(void* const* d_in, const int* in_sizes, int n_in,
                              void* d_out, int out_size, void* d_ws, size_t ws_size,
                              hipStream_t stream) {
    const float* x    = (const float*)d_in[0];
    const float* mask = (const float*)d_in[1];
    const float* W1   = (const float*)d_in[2];
    const float* b1   = (const float*)d_in[3];
    const float* W2   = (const float*)d_in[4];
    const float* b2   = (const float*)d_in[5];
    float* out = (float*)d_out;

    const int B = in_sizes[0] / (C_IN * F_DIM * NNODE);   // 16
    dim3 grid(F_DIM, B, NH / HPB);
    dim3 block(64);
    gat_fused_kernel<<<grid, block, 0, stream>>>(x, mask, W1, b1, W2, b2, out);
}
